// InteractionNetwork_28656021799343
// MI455X (gfx1250) — matrix operand order repaired and checked by reading, not compiled, not run on hardware
//
#include <hip/hip_runtime.h>

#define NN 50000
#define NE 800000
#define DIM 128

typedef __bf16 v16bf __attribute__((ext_vector_type(16)));
typedef float  v8f   __attribute__((ext_vector_type(8)));
typedef unsigned int v2u __attribute__((ext_vector_type(2)));
typedef unsigned int v4u __attribute__((ext_vector_type(4)));

struct v4u2 { v4u a, b; };   // 32B, bit-castable to v16bf

__device__ __forceinline__ unsigned short f2bf(float f) {
  unsigned int u = __builtin_bit_cast(unsigned int, f);
  u += 0x7FFFu + ((u >> 16) & 1u);           // round-to-nearest-even
  return (unsigned short)(u >> 16);
}
__device__ __forceinline__ unsigned int pack2(float lo, float hi) {
  return (unsigned int)f2bf(lo) | ((unsigned int)f2bf(hi) << 16);
}

// A fragment 16x32 bf16 from LDS. Rows m0..m0+15, k in [kb, kb+32).
// ISA layout: lane L -> row m0+(L&15); hk=(L>=16)?8:0; VGPRs 0-3: k=hk+2v(+1); 4-7: k=16+hk+2v(+1)
// Dwords are contiguous runs base+0..3 and base+8..11 -> two ds_load_b128 (base is 16B aligned).
__device__ __forceinline__ v16bf frag_a(const unsigned int* lds, int m0, int stride_dw, int kb, int lane) {
  int m = m0 + (lane & 15);
  int hk = (lane >> 4) << 3;
  int base = m * stride_dw + ((kb + hk) >> 1);     // multiple of 4 dwords
  const v4u* p = (const v4u*)(lds + base);
  v4u2 t;
  t.a = p[0];
  t.b = p[2];
  return __builtin_bit_cast(v16bf, t);
}

// B fragment 32x16 bf16 from LDS holding W^T as [n][k]. lane L -> col n0+(L&15);
// same k order as A: hk=(L>=16)?8:0; elems 0-7: k=kb+hk+s; 8-15: k=kb+16+hk+(s-8) -> dwords base+0..3, base+8..11.
__device__ __forceinline__ v16bf frag_b(const unsigned int* lds, int n0, int stride_dw, int kb, int lane) {
  int n = n0 + (lane & 15);
  int base = n * stride_dw + ((kb + ((lane >> 4) << 3)) >> 1);  // multiple of 4 dwords
  const v4u* p = (const v4u*)(lds + base);
  v4u2 t;
  t.a = p[0];
  t.b = p[2];
  return __builtin_bit_cast(v16bf, t);
}

__device__ __forceinline__ v8f wmma_bf16(v16bf a, v16bf b, v8f c) {
  return __builtin_amdgcn_wmma_f32_16x16x32_bf16(false, a, false, b, (short)0, c, false, false);
}

// -------- LDS layout (dwords; all section offsets & strides are multiples of 4) --------
#define E_SW1    0
#define E_SW1_DW (128*196)
#define E_SW2    (E_SW1_DW)
#define E_SW2_DW (128*68)
#define E_SIN    (E_SW2 + E_SW2_DW)
#define E_SIN_DW (64*196)
#define E_SH     (E_SIN + E_SIN_DW)
#define E_SH_DW  (64*68)
#define E_SMEM_DW (E_SH + E_SH_DW)        // 50688 dw = 202,752 B (< 320KB LDS/WGP)

#define Nk_SW1    0
#define Nk_SW1_DW (128*132)
#define Nk_SW2    (Nk_SW1_DW)
#define Nk_SW2_DW (128*68)
#define Nk_SIN    (Nk_SW2 + Nk_SW2_DW)
#define Nk_SIN_DW (64*132)
#define Nk_SH     (Nk_SIN + Nk_SIN_DW)
#define Nk_SH_DW  (64*68)
#define Nk_SMEM_DW (Nk_SH + Nk_SH_DW)     // 38400 dw = 153,600 B

// -------- prep: W [K][N] fp32 -> Wt [N][K] bf16 --------
__global__ void prep_wt_kernel(const float* __restrict__ W, unsigned short* __restrict__ Wt,
                               int K, int N) {
  int idx = blockIdx.x * blockDim.x + threadIdx.x;
  if (idx >= K * N) return;
  int n = idx / K, k = idx - n * K;
  Wt[(size_t)n * K + k] = f2bf(W[(size_t)k * N + n]);
}

__global__ void zero_kernel(float* __restrict__ p, int n) {
  int i = blockIdx.x * blockDim.x + threadIdx.x;
  if (i < n) p[i] = 0.0f;
}

// -------- edge MLP + scatter-add --------
__global__ void __launch_bounds__(256)
edge_mlp_kernel(const float* __restrict__ x,
                const long long* __restrict__ ei,     // [2][NE]
                const float* __restrict__ eattr,
                const float* __restrict__ be1,
                const float* __restrict__ be2,
                const unsigned int* __restrict__ w1t, // bf16 [128][384] as 192 dw/row
                const unsigned int* __restrict__ w2t, // bf16 [128][128] as  64 dw/row
                float* __restrict__ out_edge,
                float* __restrict__ agg) {
  extern __shared__ unsigned int smem[];
  const int tid = threadIdx.x;
  const int lane = tid & 31;
  const int wave = tid >> 5;
  const int e0 = blockIdx.x * 64;

  // stage weights via b128 loads / b128 LDS stores
  {
    v4u* s4 = (v4u*)smem;
    const v4u* w1q = (const v4u*)w1t;                 // 48 quads per 192-dw row
    for (int p = tid; p < 128 * 48; p += 256) {
      int n = p / 48, q = p - n * 48;
      s4[n * 49 + q] = w1q[p];                        // dst stride 196 dw = 49 quads
    }
    const v4u* w2q = (const v4u*)w2t;                 // 16 quads per 64-dw row
    for (int p = tid; p < 128 * 16; p += 256) {
      int n = p >> 4, q = p & 15;
      s4[(E_SW2 >> 2) + n * 17 + q] = w2q[p];         // dst stride 68 dw = 17 quads
    }
  }
  // gather + convert edge inputs: [x[src] | x[dst] | e_attr] -> bf16 tile 64x384
  for (int p = tid; p < 64 * 96; p += 256) {          // 96 float4 per row
    int row = p / 96, qc = p - row * 96;
    int col = qc << 2;
    int e = e0 + row;
    float4 f;
    if (col < 128) {
      int s = (int)ei[e];
      f = *(const float4*)(x + (size_t)s * DIM + col);
    } else if (col < 256) {
      int d = (int)ei[NE + e];
      f = *(const float4*)(x + (size_t)d * DIM + (col - 128));
    } else {
      f = *(const float4*)(eattr + (size_t)e * DIM + (col - 256));
    }
    v2u pk;
    pk.x = pack2(f.x, f.y);
    pk.y = pack2(f.z, f.w);
    *(v2u*)(smem + E_SIN + row * 196 + (qc << 1)) = pk;  // ds_store_b64
  }
  __syncthreads();

  const int n0 = wave << 4;              // 8 waves cover N=128
  const int nn = n0 + (lane & 15);
  const int mr = (lane >> 4) << 3;       // C/D: rows r (lanes 0-15) and r+8 (lanes 16-31)
  const float b1 = be1[nn];
  const float b2 = be2[nn];
  unsigned short* sh16 = (unsigned short*)(smem + E_SH);

  // layer 1: h = relu(in @ We1 + be1), K=384 -> 12 WMMA per 16x16 tile
#pragma unroll
  for (int mt = 0; mt < 4; ++mt) {
    int m0 = mt << 4;
    v8f c = {};
#pragma unroll
    for (int kt = 0; kt < 12; ++kt) {
      v16bf a = frag_a(smem + E_SIN, m0, 196, kt * 32, lane);
      v16bf b = frag_b(smem + E_SW1, n0, 196, kt * 32, lane);
      c = wmma_bf16(a, b, c);
    }
#pragma unroll
    for (int r = 0; r < 8; ++r) {
      float v = c[r] + b1;
      v = v > 0.0f ? v : 0.0f;
      sh16[(m0 + mr + r) * 136 + nn] = f2bf(v);
    }
  }
  __syncthreads();

  // layer 2: updated_edge = h @ We2 + be2, then scatter-add to agg[dest]
#pragma unroll
  for (int mt = 0; mt < 4; ++mt) {
    int m0 = mt << 4;
    v8f c = {};
#pragma unroll
    for (int kt = 0; kt < 4; ++kt) {
      v16bf a = frag_a(smem + E_SH, m0, 68, kt * 32, lane);
      v16bf b = frag_b(smem + E_SW2, n0, 68, kt * 32, lane);
      c = wmma_bf16(a, b, c);
    }
#pragma unroll
    for (int r = 0; r < 8; ++r) {
      int m = m0 + mr + r;
      int e = e0 + m;
      float v = c[r] + b2;
      out_edge[(size_t)e * DIM + nn] = v;
      int d = (int)ei[NE + e];
      atomicAdd(agg + (size_t)d * DIM + nn, v);   // global_atomic_add_f32, L2-resident target
    }
  }
}

// -------- node MLP --------
__global__ void __launch_bounds__(256)
node_mlp_kernel(const float* __restrict__ x,
                const float* __restrict__ agg,
                const float* __restrict__ bn1,
                const float* __restrict__ bn2,
                const unsigned int* __restrict__ w1t, // bf16 [128][256] as 128 dw/row
                const unsigned int* __restrict__ w2t, // bf16 [128][128] as  64 dw/row
                float* __restrict__ out_node) {
  extern __shared__ unsigned int smem[];
  const int tid = threadIdx.x;
  const int lane = tid & 31;
  const int wave = tid >> 5;
  const int i0 = blockIdx.x * 64;

  {
    v4u* s4 = (v4u*)smem;
    const v4u* w1q = (const v4u*)w1t;                 // 32 quads per 128-dw row
    for (int p = tid; p < 128 * 32; p += 256) {
      int n = p >> 5, q = p & 31;
      s4[n * 33 + q] = w1q[p];                        // dst stride 132 dw = 33 quads
    }
    const v4u* w2q = (const v4u*)w2t;
    for (int p = tid; p < 128 * 16; p += 256) {
      int n = p >> 4, q = p & 15;
      s4[(Nk_SW2 >> 2) + n * 17 + q] = w2q[p];
    }
  }
  for (int p = tid; p < 64 * 64; p += 256) {          // 64 float4 per 256-elem row
    int row = p >> 6, qc = p & 63;
    int col = qc << 2;
    int i = i0 + row;
    if (i >= NN) i = NN - 1;               // clamp tail: WMMA needs full tiles
    float4 f = (col < 128) ? *(const float4*)(x   + (size_t)i * DIM + col)
                           : *(const float4*)(agg + (size_t)i * DIM + (col - 128));
    v2u pk;
    pk.x = pack2(f.x, f.y);
    pk.y = pack2(f.z, f.w);
    *(v2u*)(smem + Nk_SIN + row * 132 + (qc << 1)) = pk;
  }
  __syncthreads();

  const int n0 = wave << 4;
  const int nn = n0 + (lane & 15);
  const int mr = (lane >> 4) << 3;
  const float b1 = bn1[nn];
  const float b2 = bn2[nn];
  unsigned short* sh16 = (unsigned short*)(smem + Nk_SH);

#pragma unroll
  for (int mt = 0; mt < 4; ++mt) {
    int m0 = mt << 4;
    v8f c = {};
#pragma unroll
    for (int kt = 0; kt < 8; ++kt) {       // K = 256
      v16bf a = frag_a(smem + Nk_SIN, m0, 132, kt * 32, lane);
      v16bf b = frag_b(smem + Nk_SW1, n0, 132, kt * 32, lane);
      c = wmma_bf16(a, b, c);
    }
#pragma unroll
    for (int r = 0; r < 8; ++r) {
      float v = c[r] + b1;
      v = v > 0.0f ? v : 0.0f;
      sh16[(m0 + mr + r) * 136 + nn] = f2bf(v);
    }
  }
  __syncthreads();

#pragma unroll
  for (int mt = 0; mt < 4; ++mt) {
    int m0 = mt << 4;
    v8f c = {};
#pragma unroll
    for (int kt = 0; kt < 4; ++kt) {       // K = 128
      v16bf a = frag_a(smem + Nk_SH, m0, 68, kt * 32, lane);
      v16bf b = frag_b(smem + Nk_SW2, n0, 68, kt * 32, lane);
      c = wmma_bf16(a, b, c);
    }
#pragma unroll
    for (int r = 0; r < 8; ++r) {
      int m = i0 + m0 + mr + r;
      if (m < NN) out_node[(size_t)m * DIM + nn] = c[r] + b2;
    }
  }
}

extern "C" void kernel_launch(void* const* d_in, const int* in_sizes, int n_in,
                              void* d_out, int out_size, void* d_ws, size_t ws_size,
                              hipStream_t stream) {
  (void)in_sizes; (void)n_in; (void)out_size; (void)ws_size;
  const float*     x     = (const float*)d_in[0];
  const long long* ei    = (const long long*)d_in[1];
  const float*     eattr = (const float*)d_in[2];
  const float*     We1   = (const float*)d_in[3];
  const float*     be1   = (const float*)d_in[4];
  const float*     We2   = (const float*)d_in[5];
  const float*     be2   = (const float*)d_in[6];
  const float*     Wn1   = (const float*)d_in[7];
  const float*     bn1   = (const float*)d_in[8];
  const float*     Wn2   = (const float*)d_in[9];
  const float*     bn2   = (const float*)d_in[10];

  char* ws = (char*)d_ws;
  unsigned short* We1t = (unsigned short*)(ws);            // 384*128*2 = 98304 B
  unsigned short* We2t = (unsigned short*)(ws + 98304);    // 32768 B
  unsigned short* Wn1t = (unsigned short*)(ws + 131072);   // 65536 B
  unsigned short* Wn2t = (unsigned short*)(ws + 196608);   // 32768 B
  float*          agg  = (float*)(ws + 229376);            // 50000*128*4 = 25.6 MB

  float* out_node = (float*)d_out;
  float* out_edge = out_node + (size_t)NN * DIM;

  (void)hipFuncSetAttribute((const void*)edge_mlp_kernel,
                            hipFuncAttributeMaxDynamicSharedMemorySize, E_SMEM_DW * 4);
  (void)hipFuncSetAttribute((const void*)node_mlp_kernel,
                            hipFuncAttributeMaxDynamicSharedMemorySize, Nk_SMEM_DW * 4);

  prep_wt_kernel<<<(384 * 128 + 255) / 256, 256, 0, stream>>>(We1, We1t, 384, 128);
  prep_wt_kernel<<<(128 * 128 + 255) / 256, 256, 0, stream>>>(We2, We2t, 128, 128);
  prep_wt_kernel<<<(256 * 128 + 255) / 256, 256, 0, stream>>>(Wn1, Wn1t, 256, 128);
  prep_wt_kernel<<<(128 * 128 + 255) / 256, 256, 0, stream>>>(Wn2, Wn2t, 128, 128);
  zero_kernel<<<(NN * DIM + 255) / 256, 256, 0, stream>>>(agg, NN * DIM);

  edge_mlp_kernel<<<NE / 64, 256, E_SMEM_DW * 4, stream>>>(
      x, ei, eattr, be1, be2,
      (const unsigned int*)We1t, (const unsigned int*)We2t, out_edge, agg);
  node_mlp_kernel<<<(NN + 63) / 64, 256, Nk_SMEM_DW * 4, stream>>>(
      x, agg, bn1, bn2,
      (const unsigned int*)Wn1t, (const unsigned int*)Wn2t, out_node);
}
